// RadarOccupancyModel_34720515621000
// MI455X (gfx1250) — matrix operand order repaired and checked by reading, not compiled, not run on hardware
//
#include <hip/hip_runtime.h>
#include <hip/hip_bf16.h>
#include <math.h>
#include <stdint.h>

// ---------------------------------------------------------------------------
// RadarOccupancyModel forward for MI455X (gfx1250, wave32, WMMA).
// All einsum layers run on v_wmma_f32_16x16x32_f16 with f32 accumulation.
// BN is two-pass (stats pass + fused norm/relu[/max16] pass), recomputing the
// GEMM instead of materializing raw activations (~35 GFLOP extra vs ~80 MB
// saved HBM traffic; WMMA makes the FLOPs free at 23.3 TB/s memory).
// GEMM tiles are staged into LDS with gfx1250 async global->LDS DMA
// (ASYNCcnt) and double-buffered so copy overlaps the WMMA pipeline.
// ---------------------------------------------------------------------------

typedef __attribute__((ext_vector_type(16))) _Float16 v16h;
typedef __attribute__((ext_vector_type(8)))  _Float16 v8h;
typedef __attribute__((ext_vector_type(8)))  float    v8f;
typedef __attribute__((ext_vector_type(4)))  float    v4f;
typedef __attribute__((ext_vector_type(4)))  int      v4i;

#define NTOT   4720      // A*E*R = 59*8*10
#define NRET   2360
#define BATCH  32
#define S1P    1180
#define S2P    295
#define S3P    59

// ---- gfx1250 async global->LDS copy (ASYNCcnt) -----------------------------
#if defined(__has_builtin)
# if __has_builtin(__builtin_amdgcn_global_load_async_to_lds_b128) && \
     __has_builtin(__builtin_amdgcn_s_wait_asynccnt)
#  define HAVE_ASYNC_LDS 1
# endif
#endif
#ifndef HAVE_ASYNC_LDS
# define HAVE_ASYNC_LDS 0
#endif

#if HAVE_ASYNC_LDS
typedef __attribute__((address_space(1))) v4i as1_v4i;   // global int4
typedef __attribute__((address_space(3))) v4i as3_v4i;   // LDS int4
// Generic->AS1: global VA is the 64-bit value itself.
// Generic->AS3: generic LDS pointers are {aperture, offset32}; low 32 bits are
// the DS byte offset.
__device__ __forceinline__ void async_copy_b128(const void* g, void* l) {
    __builtin_amdgcn_global_load_async_to_lds_b128(
        (as1_v4i*)(uintptr_t)g,
        (as3_v4i*)(unsigned int)(uintptr_t)l, 0, 0);
}
#endif

// ---------------------------------------------------------------------------
// 1) Polar grid -> cartesian points (A,R,E order flattened)
// ---------------------------------------------------------------------------
__global__ void points_kernel(const float* __restrict__ az_s, const float* __restrict__ az_b,
                              const float* __restrict__ az_cw, const float* __restrict__ az_sw,
                              const float* __restrict__ el_s, const float* __restrict__ el_b,
                              const float* __restrict__ el_cw, const float* __restrict__ el_sw,
                              const float* __restrict__ r_s, const float* __restrict__ r_b,
                              float* __restrict__ pts) {
    int i = blockIdx.x * blockDim.x + threadIdx.x;
    if (i >= NTOT) return;
    int a = i / 80;          // R*E = 80
    int rem = i % 80;
    int r = rem / 8;
    int e = rem % 8;
    float azbin = -1.0472f + 2.0944f * (float)a / 58.f;
    float az    = azbin * az_s[a] + az_b[a];
    float ca    = az_cw[a] * cosf(az);
    float sa    = az_sw[a] * sinf(az);
    float elbin = -0.2618f + 0.5236f * (float)e / 7.f;
    float el    = elbin * el_s[e] + el_b[e];
    float ce    = el_cw[e] * cosf(el);
    float se    = el_sw[e] * sinf(el);
    float rg    = (5.f / 9.f) * (float)r * r_s[r] + r_b[r];
    pts[i * 3 + 0] = rg * ce * ca;
    pts[i * 3 + 1] = rg * ce * sa;
    pts[i * 3 + 2] = rg * se;
}

// ---------------------------------------------------------------------------
// 2) top_k(sigmoid(dropout_w), NRET) -> sorted indices (rank by value, ties
//    broken by lower index; sigmoid is monotone so rank dropout_w directly)
// ---------------------------------------------------------------------------
__global__ void topk_kernel(const float* __restrict__ dw, int* __restrict__ keep,
                            int n, int k) {
    __shared__ unsigned char flag[NTOT];
    int tid = threadIdx.x;
    for (int i = tid; i < n; i += blockDim.x) {
        float vi = dw[i];
        int rank = 0;
        for (int j = 0; j < n; ++j) {
            float vj = dw[j];
            rank += (vj > vi) || (vj == vi && j < i);
        }
        flag[i] = (rank < k) ? 1 : 0;
    }
    __syncthreads();
    if (tid == 0) {
        int c = 0;
        for (int i = 0; i < n; ++i)
            if (flag[i]) keep[c++] = i;
    }
}

// ---------------------------------------------------------------------------
// 3) Gather retained cloud: xyz0[b][n][3], f0[b][n] (intensity)
// ---------------------------------------------------------------------------
__global__ void cloud_kernel(const float* __restrict__ pts, const float* __restrict__ polar,
                             const int* __restrict__ keep, float* __restrict__ xyz0,
                             float* __restrict__ f0, int B, int Nr, int Ntot) {
    int id = blockIdx.x * blockDim.x + threadIdx.x;
    if (id >= B * Nr) return;
    int b = id / Nr, n = id % Nr;
    int k = keep[n];
    xyz0[id * 3 + 0] = pts[k * 3 + 0];
    xyz0[id * 3 + 1] = pts[k * 3 + 1];
    xyz0[id * 3 + 2] = pts[k * 3 + 2];
    f0[id] = polar[(size_t)b * Ntot + k];
}

// ---------------------------------------------------------------------------
// 4) Farthest point sampling: one block per batch; matches jax.lax.scan
//    semantics (first sample index 0, argmax ties -> lowest index)
// ---------------------------------------------------------------------------
__global__ void fps_kernel(const float* __restrict__ xyz, float* __restrict__ nxyz,
                           int N, int S) {
    __shared__ float dist[NRET];     // N <= 2360
    __shared__ float rv[256];
    __shared__ int   ri[256];
    __shared__ int   cur;
    const int b = blockIdx.x;
    const int tid = threadIdx.x;
    const float* P = xyz + (size_t)b * N * 3;
    for (int i = tid; i < N; i += 256) dist[i] = 1e10f;
    if (tid == 0) cur = 0;
    __syncthreads();
    for (int it = 0; it < S; ++it) {
        int far = cur;
        float cx = P[far * 3 + 0], cy = P[far * 3 + 1], cz = P[far * 3 + 2];
        if (tid == 0) {
            float* o = nxyz + ((size_t)b * S + it) * 3;
            o[0] = cx; o[1] = cy; o[2] = cz;
        }
        float bv = -1.f; int bi = 0;
        for (int i = tid; i < N; i += 256) {
            float dx = P[i * 3 + 0] - cx, dy = P[i * 3 + 1] - cy, dz = P[i * 3 + 2] - cz;
            float d  = dx * dx + dy * dy + dz * dz;
            float nd = fminf(dist[i], d);
            dist[i]  = nd;
            if (nd > bv) { bv = nd; bi = i; }
        }
        rv[tid] = bv; ri[tid] = bi;
        __syncthreads();
        for (int off = 128; off > 0; off >>= 1) {
            if (tid < off) {
                float ov = rv[tid + off]; int oi = ri[tid + off];
                if (ov > rv[tid] || (ov == rv[tid] && oi < ri[tid])) { rv[tid] = ov; ri[tid] = oi; }
            }
            __syncthreads();
        }
        if (tid == 0) cur = ri[0];
        __syncthreads();
    }
}

// ---------------------------------------------------------------------------
// 5) Ball query (16 smallest in-radius indices, padded with first) + grouping
//    grouped[b][s][k][0:3] = xyz[j]-q ; [3:3+Cf] = feats[j]
// ---------------------------------------------------------------------------
__global__ void group_kernel(const float* __restrict__ xyz, const float* __restrict__ nxyz,
                             const float* __restrict__ feats, float* __restrict__ grouped,
                             int B, int N, int S, int Cf, float r2) {
    int id = blockIdx.x * blockDim.x + threadIdx.x;
    if (id >= B * S) return;
    int b = id / S;
    const float* P = xyz + (size_t)b * N * 3;
    const float* q = nxyz + (size_t)id * 3;
    float qx = q[0], qy = q[1], qz = q[2];
    int idxs[16]; int cnt = 0;
    for (int j = 0; j < N && cnt < 16; ++j) {
        float dx = P[j * 3 + 0] - qx, dy = P[j * 3 + 1] - qy, dz = P[j * 3 + 2] - qz;
        if (dx * dx + dy * dy + dz * dz <= r2) idxs[cnt++] = j;
    }
    if (cnt == 0) { idxs[0] = 0; cnt = 1; }
    for (int k = cnt; k < 16; ++k) idxs[k] = idxs[0];
    int C = 3 + Cf;
    float* g = grouped + (size_t)id * 16 * C;
    for (int k = 0; k < 16; ++k) {
        int j = idxs[k];
        float* gr = g + k * C;
        gr[0] = P[j * 3 + 0] - qx;
        gr[1] = P[j * 3 + 1] - qy;
        gr[2] = P[j * 3 + 2] - qz;
        const float* f = feats + ((size_t)b * N + j) * Cf;
        for (int c = 0; c < Cf; ++c) gr[3 + c] = f[c];
    }
}

// ---------------------------------------------------------------------------
// 6) WMMA GEMM with fused BN epilogues.
//    Fragments per ISA 7.12.2:
//      A 16x32 f16: M = lane&15; halves 0-7 -> K=kb..kb+7, 8-15 -> K=kb+16..kb+23
//      B 32x16 f16: N = lane&15; same K order as A
//    LDS tiles held as f32; f32->f16 conversion fused into fragment build.
// ---------------------------------------------------------------------------
__device__ __forceinline__ v8f tile_mma(const float* As, const float* Ws,
                                        int wave, int lane, v8f acc) {
    const int mrow = (wave << 4) + (lane & 15);
    const int kbA  = (lane >= 16) ? 8 : 0;
    const float* Ap = As + mrow * 32 + kbA;
    v4f a0 = *(const v4f*)(Ap);
    v4f a1 = *(const v4f*)(Ap + 4);
    v4f a2 = *(const v4f*)(Ap + 16);
    v4f a3 = *(const v4f*)(Ap + 20);
    const int nb  = lane & 15;
    const int kbB = (lane >= 16) ? 8 : 0;
    const float* Wp = Ws + nb * 32 + kbB;
    v4f b0 = *(const v4f*)(Wp);
    v4f b1 = *(const v4f*)(Wp + 4);
    v4f b2 = *(const v4f*)(Wp + 16);
    v4f b3 = *(const v4f*)(Wp + 20);
    v16h a, b;
    for (int i = 0; i < 4; ++i) {
        a[i]      = (_Float16)a0[i];
        a[i + 4]  = (_Float16)a1[i];
        a[i + 8]  = (_Float16)a2[i];
        a[i + 12] = (_Float16)a3[i];
        b[i]      = (_Float16)b0[i];
        b[i + 4]  = (_Float16)b1[i];
        b[i + 8]  = (_Float16)b2[i];
        b[i + 12] = (_Float16)b3[i];
    }
    return __builtin_amdgcn_wmma_f32_16x16x32_f16(false, a, false, b, (short)0,
                                                  acc, false, false);
}

__global__ void wmma_bn_gemm(const float* __restrict__ A, const float* __restrict__ Wt,
                             const float* __restrict__ bias, int M, int K, int N,
                             float* __restrict__ out, float* __restrict__ stats, int mode) {
    __shared__ __align__(16) float As[2][64 * 32];   // 16 KB
    __shared__ __align__(16) float Ws[2][16 * 32];   //  4 KB
    __shared__ float s_sum[16];
    __shared__ float s_sq[16];
    const int tid  = threadIdx.x;
    const int wave = tid >> 5;
    const int lane = tid & 31;
    const int m0 = blockIdx.x * 64;
    const int n0 = blockIdx.y * 16;
    v8f acc = {};
#if HAVE_ASYNC_LDS
    const bool fast = ((K & 31) == 0) && (m0 + 64 <= M);   // uniform per block
    if (fast) {
        // Async double-buffered staging: each wave issues exactly 5 async b128
        // copies per tile (4 for A, 1 for W); in-order ASYNCcnt completion =>
        // wait<=5 after issuing tile t+1 guarantees tile t has landed.
        auto stage = [&](int k0, int buf) {
            for (int c = tid; c < 512; c += 128) {            // A: 64x32 f32
                int row = c >> 3, seg = (c & 7) << 2;
                async_copy_b128(A + (size_t)(m0 + row) * K + k0 + seg,
                                &As[buf][row * 32 + seg]);
            }
            int row = tid >> 3, seg = (tid & 7) << 2;          // W: 16x32 f32
            async_copy_b128(Wt + (size_t)(n0 + row) * K + k0 + seg,
                            &Ws[buf][row * 32 + seg]);
        };
        const int nk = K >> 5;
        stage(0, 0);
        for (int t = 0; t < nk; ++t) {
            if (t + 1 < nk) {
                stage(t + 1, (t + 1) & 1);
                __builtin_amdgcn_s_wait_asynccnt(5);
            } else {
                __builtin_amdgcn_s_wait_asynccnt(0);
            }
            __syncthreads();                 // all waves waited their own chunks
            acc = tile_mma(&As[t & 1][0], &Ws[t & 1][0], wave, lane, acc);
            __syncthreads();                 // protect buffer reuse at t+2
        }
    } else
#endif
    {
        for (int k0 = 0; k0 < K; k0 += 32) {
            for (int e = tid; e < 64 * 32; e += 128) {
                int m = e >> 5, k = e & 31;
                int gm = m0 + m, gk = k0 + k;
                As[0][e] = (gm < M && gk < K) ? A[(size_t)gm * K + gk] : 0.f;
            }
            for (int e = tid; e < 16 * 32; e += 128) {
                int n = e >> 5, k = e & 31;
                int gk = k0 + k;
                Ws[0][e] = (gk < K) ? Wt[(size_t)(n0 + n) * K + gk] : 0.f;
            }
            __syncthreads();
            acc = tile_mma(&As[0][0], &Ws[0][0], wave, lane, acc);
            __syncthreads();
        }
    }
    const int n  = n0 + (lane & 15);
    const float bv = bias[n];
    const int mbase = m0 + (wave << 4) + ((lane >= 16) ? 8 : 0);
    if (mode == 0) {
        float s = 0.f, s2 = 0.f;
        for (int r = 0; r < 8; ++r) {
            if (mbase + r < M) {
                float y = acc[r] + bv;
                s += y; s2 += y * y;
            }
        }
        if (tid < 16) { s_sum[tid] = 0.f; s_sq[tid] = 0.f; }
        __syncthreads();
        atomicAdd(&s_sum[lane & 15], s);
        atomicAdd(&s_sq[lane & 15], s2);
        __syncthreads();
        if (tid < 16) {
            atomicAdd(&stats[n0 + tid], s_sum[tid]);
            atomicAdd(&stats[N + n0 + tid], s_sq[tid]);
        }
    } else {
        const float inv  = 1.f / (float)M;
        const float mean = stats[n] * inv;
        const float var  = stats[N + n] * inv - mean * mean;
        const float sc   = rsqrtf(var + 1e-5f);
        if (mode == 1) {
            for (int r = 0; r < 8; ++r) {
                int m = mbase + r;
                if (m < M) {
                    float y = (acc[r] + bv - mean) * sc;
                    out[(size_t)m * N + n] = fmaxf(y, 0.f);
                }
            }
        } else { // mode 2: max over the 16-row sample group (tile-aligned)
            float mx = -1e30f;
            for (int r = 0; r < 8; ++r) {
                float y = (acc[r] + bv - mean) * sc;
                mx = fmaxf(mx, fmaxf(y, 0.f));
            }
            mx = fmaxf(mx, __shfl_xor(mx, 16, 32));   // combine rows 0-7 with 8-15
            const int g = (m0 >> 4) + wave;
            if (lane < 16 && g < (M >> 4)) out[(size_t)g * N + n] = mx;
        }
    }
}

// ---------------------------------------------------------------------------
// 7) Feature propagation: 3-NN inverse-distance interpolation + concat
// ---------------------------------------------------------------------------
__global__ void fp_interp_kernel(const float* __restrict__ xyz1, const float* __restrict__ xyz2,
                                 const float* __restrict__ p1, const float* __restrict__ p2,
                                 float* __restrict__ out, int B, int N1, int N2, int C1, int C2) {
    int id = blockIdx.x * blockDim.x + threadIdx.x;
    if (id >= B * N1) return;
    int b = id / N1;
    const float* q = xyz1 + (size_t)id * 3;
    const float* P = xyz2 + (size_t)b * N2 * 3;
    float qx = q[0], qy = q[1], qz = q[2];
    float d0 = 1e30f, d1 = 1e30f, d2 = 1e30f;
    int   i0 = 0, i1 = 0, i2 = 0;
    for (int j = 0; j < N2; ++j) {
        float dx = P[j * 3 + 0] - qx, dy = P[j * 3 + 1] - qy, dz = P[j * 3 + 2] - qz;
        float d = dx * dx + dy * dy + dz * dz;
        if (d < d0)      { d2 = d1; i2 = i1; d1 = d0; i1 = i0; d0 = d; i0 = j; }
        else if (d < d1) { d2 = d1; i2 = i1; d1 = d;  i1 = j; }
        else if (d < d2) { d2 = d;  i2 = j; }
    }
    float w0 = 1.f / (d0 + 1e-8f), w1 = 1.f / (d1 + 1e-8f), w2 = 1.f / (d2 + 1e-8f);
    float wsum = w0 + w1 + w2;
    w0 /= wsum; w1 /= wsum; w2 /= wsum;
    const int C = C1 + C2;
    float* o = out + (size_t)id * C;
    if (C1 > 0) {
        const float* f1 = p1 + (size_t)id * C1;
        for (int c = 0; c < C1; ++c) o[c] = f1[c];
    }
    const float* base = p2 + (size_t)b * N2 * C2;
    const float* fa = base + (size_t)i0 * C2;
    const float* fb = base + (size_t)i1 * C2;
    const float* fc = base + (size_t)i2 * C2;
    for (int c = 0; c < C2; ++c)
        o[C1 + c] = w0 * fa[c] + w1 * fb[c] + w2 * fc[c];
}

// ---------------------------------------------------------------------------
// 8) Head2: 128 -> 1 dot product + sigmoid
// ---------------------------------------------------------------------------
__global__ void head2_kernel(const float* __restrict__ h, const float* __restrict__ w,
                             const float* __restrict__ b2, float* __restrict__ out, int total) {
    int id = blockIdx.x * blockDim.x + threadIdx.x;
    if (id >= total) return;
    const float* x = h + (size_t)id * 128;
    float acc = b2[0];
    for (int c = 0; c < 128; ++c) acc += x[c] * w[c];
    out[id] = 1.f / (1.f + expf(-acc));
}

// ---------------------------------------------------------------------------
// Host orchestration
// ---------------------------------------------------------------------------
static void bn_layer(const float* in, const float* w, const float* b, int M, int K, int N,
                     float* out, float* stats, bool maxpool, hipStream_t s) {
    (void)hipMemsetAsync(stats, 0, 2 * N * sizeof(float), s);
    dim3 grid((M + 63) / 64, N / 16);
    wmma_bn_gemm<<<grid, 128, 0, s>>>(in, w, b, M, K, N, nullptr, stats, 0);
    wmma_bn_gemm<<<grid, 128, 0, s>>>(in, w, b, M, K, N, out, stats, maxpool ? 2 : 1);
}

extern "C" void kernel_launch(void* const* d_in, const int* in_sizes, int n_in,
                              void* d_out, int out_size, void* d_ws, size_t ws_size,
                              hipStream_t stream) {
    // Input order: polar_frames, then params pytree leaves (dict keys sorted):
    // az_bias, az_cos_w, az_scale, az_sin_w, dropout_w, el_bias, el_cos_w,
    // el_scale, el_sin_w, fp1(w,b x3), fp2(w,b x2), fp3(w,b x2),
    // head1(w,b), head2(w,b), r_bias, r_scale, sa1(w,b x3), sa2(w,b x3), sa3(w,b x3)
    const float* polar   = (const float*)d_in[0];
    const float* az_bias = (const float*)d_in[1];
    const float* az_cw   = (const float*)d_in[2];
    const float* az_sc   = (const float*)d_in[3];
    const float* az_sw   = (const float*)d_in[4];
    const float* drop_w  = (const float*)d_in[5];
    const float* el_bias = (const float*)d_in[6];
    const float* el_cw   = (const float*)d_in[7];
    const float* el_sc   = (const float*)d_in[8];
    const float* el_sw   = (const float*)d_in[9];
    const float* fp1w0 = (const float*)d_in[10]; const float* fp1b0 = (const float*)d_in[11];
    const float* fp1w1 = (const float*)d_in[12]; const float* fp1b1 = (const float*)d_in[13];
    const float* fp1w2 = (const float*)d_in[14]; const float* fp1b2 = (const float*)d_in[15];
    const float* fp2w0 = (const float*)d_in[16]; const float* fp2b0 = (const float*)d_in[17];
    const float* fp2w1 = (const float*)d_in[18]; const float* fp2b1 = (const float*)d_in[19];
    const float* fp3w0 = (const float*)d_in[20]; const float* fp3b0 = (const float*)d_in[21];
    const float* fp3w1 = (const float*)d_in[22]; const float* fp3b1 = (const float*)d_in[23];
    const float* h1w   = (const float*)d_in[24]; const float* h1b   = (const float*)d_in[25];
    const float* h2w   = (const float*)d_in[26]; const float* h2b   = (const float*)d_in[27];
    const float* r_bias = (const float*)d_in[28];
    const float* r_sc   = (const float*)d_in[29];
    const float* sa1w0 = (const float*)d_in[30]; const float* sa1b0 = (const float*)d_in[31];
    const float* sa1w1 = (const float*)d_in[32]; const float* sa1b1 = (const float*)d_in[33];
    const float* sa1w2 = (const float*)d_in[34]; const float* sa1b2 = (const float*)d_in[35];
    const float* sa2w0 = (const float*)d_in[36]; const float* sa2b0 = (const float*)d_in[37];
    const float* sa2w1 = (const float*)d_in[38]; const float* sa2b1 = (const float*)d_in[39];
    const float* sa2w2 = (const float*)d_in[40]; const float* sa2b2 = (const float*)d_in[41];
    const float* sa3w0 = (const float*)d_in[42]; const float* sa3b0 = (const float*)d_in[43];
    const float* sa3w1 = (const float*)d_in[44]; const float* sa3b1 = (const float*)d_in[45];
    const float* sa3w2 = (const float*)d_in[46]; const float* sa3b2 = (const float*)d_in[47];
    (void)in_sizes; (void)n_in; (void)out_size; (void)ws_size;

    // Workspace layout (~173 MB total)
    char* ws = (char*)d_ws;
    size_t off = 0;
    auto allocf = [&](size_t nfl) -> float* {
        float* p = (float*)(ws + off);
        off += ((nfl * sizeof(float) + 255) / 256) * 256;
        return p;
    };
    const size_t BUFSZ = (size_t)BATCH * S1P * 16 * 32;   // 19,333,120 floats
    float* pts   = allocf((size_t)NTOT * 3);
    int*   keep  = (int*)allocf(NRET);
    float* xyz0  = allocf((size_t)BATCH * NRET * 3);
    float* f0    = allocf((size_t)BATCH * NRET);
    float* l1x   = allocf((size_t)BATCH * S1P * 3);
    float* l2x   = allocf((size_t)BATCH * S2P * 3);
    float* l3x   = allocf((size_t)BATCH * S3P * 3);
    float* l1f   = allocf((size_t)BATCH * S1P * 64);
    float* l2f   = allocf((size_t)BATCH * S2P * 128);
    float* l3f   = allocf((size_t)BATCH * S3P * 256);
    float* stats = allocf(512);
    float* bufA  = allocf(BUFSZ);
    float* bufB  = allocf(BUFSZ);

    // --- point cloud construction + dropout ---
    points_kernel<<<(NTOT + 255) / 256, 256, 0, stream>>>(az_sc, az_bias, az_cw, az_sw,
                                                          el_sc, el_bias, el_cw, el_sw,
                                                          r_sc, r_bias, pts);
    topk_kernel<<<1, 1024, 0, stream>>>(drop_w, keep, NTOT, NRET);
    cloud_kernel<<<(BATCH * NRET + 255) / 256, 256, 0, stream>>>(pts, polar, keep, xyz0, f0,
                                                                 BATCH, NRET, NTOT);
    // --- SA1: 2360 -> 1180, r=0.2, C=4, mlp 4->32->32->64 ---
    fps_kernel<<<BATCH, 256, 0, stream>>>(xyz0, l1x, NRET, S1P);
    group_kernel<<<(BATCH * S1P + 127) / 128, 128, 0, stream>>>(xyz0, l1x, f0, bufA,
                                                                BATCH, NRET, S1P, 1, 0.04f);
    bn_layer(bufA, sa1w0, sa1b0, BATCH * S1P * 16, 4, 32, bufB, stats, false, stream);
    bn_layer(bufB, sa1w1, sa1b1, BATCH * S1P * 16, 32, 32, bufA, stats, false, stream);
    bn_layer(bufA, sa1w2, sa1b2, BATCH * S1P * 16, 32, 64, l1f, stats, true, stream);
    // --- SA2: 1180 -> 295, r=0.4, C=67, mlp 67->64->64->128 ---
    fps_kernel<<<BATCH, 256, 0, stream>>>(l1x, l2x, S1P, S2P);
    group_kernel<<<(BATCH * S2P + 127) / 128, 128, 0, stream>>>(l1x, l2x, l1f, bufB,
                                                                BATCH, S1P, S2P, 64, 0.16f);
    bn_layer(bufB, sa2w0, sa2b0, BATCH * S2P * 16, 67, 64, bufA, stats, false, stream);
    bn_layer(bufA, sa2w1, sa2b1, BATCH * S2P * 16, 64, 64, bufB, stats, false, stream);
    bn_layer(bufB, sa2w2, sa2b2, BATCH * S2P * 16, 64, 128, l2f, stats, true, stream);
    // --- SA3: 295 -> 59, r=0.6, C=131, mlp 131->128->128->256 ---
    fps_kernel<<<BATCH, 256, 0, stream>>>(l2x, l3x, S2P, S3P);
    group_kernel<<<(BATCH * S3P + 127) / 128, 128, 0, stream>>>(l2x, l3x, l2f, bufA,
                                                                BATCH, S2P, S3P, 128, 0.36f);
    bn_layer(bufA, sa3w0, sa3b0, BATCH * S3P * 16, 131, 128, bufB, stats, false, stream);
    bn_layer(bufB, sa3w1, sa3b1, BATCH * S3P * 16, 128, 128, bufA, stats, false, stream);
    bn_layer(bufA, sa3w2, sa3b2, BATCH * S3P * 16, 128, 256, l3f, stats, true, stream);
    // --- FP3: interp(l2x<-l3x) concat l2f -> 384, mlp 384->256->256 ---
    fp_interp_kernel<<<(BATCH * S2P + 127) / 128, 128, 0, stream>>>(l2x, l3x, l2f, l3f, bufB,
                                                                    BATCH, S2P, S3P, 128, 256);
    bn_layer(bufB, fp3w0, fp3b0, BATCH * S2P, 384, 256, bufA, stats, false, stream);
    bn_layer(bufA, fp3w1, fp3b1, BATCH * S2P, 256, 256, bufB, stats, false, stream);  // fp3 out in bufB
    // --- FP2: interp(l1x<-l2x) concat l1f -> 320, mlp 320->256->128 ---
    fp_interp_kernel<<<(BATCH * S1P + 127) / 128, 128, 0, stream>>>(l1x, l2x, l1f, bufB, bufA,
                                                                    BATCH, S1P, S2P, 64, 256);
    bn_layer(bufA, fp2w0, fp2b0, BATCH * S1P, 320, 256, bufB, stats, false, stream);
    bn_layer(bufB, fp2w1, fp2b1, BATCH * S1P, 256, 128, bufA, stats, false, stream);  // fp2 out in bufA
    // --- FP1: interp(xyz0<-l1x) -> 128, mlp 128->128->128->128 ---
    fp_interp_kernel<<<(BATCH * NRET + 127) / 128, 128, 0, stream>>>(xyz0, l1x, nullptr, bufA, bufB,
                                                                     BATCH, NRET, S1P, 0, 128);
    bn_layer(bufB, fp1w0, fp1b0, BATCH * NRET, 128, 128, bufA, stats, false, stream);
    bn_layer(bufA, fp1w1, fp1b1, BATCH * NRET, 128, 128, bufB, stats, false, stream);
    bn_layer(bufB, fp1w2, fp1b2, BATCH * NRET, 128, 128, bufA, stats, false, stream); // l0f in bufA
    // --- heads ---
    bn_layer(bufA, h1w, h1b, BATCH * NRET, 128, 128, bufB, stats, false, stream);
    head2_kernel<<<(BATCH * NRET + 255) / 256, 256, 0, stream>>>(bufB, h2w, h2b,
                                                                 (float*)d_out, BATCH * NRET);
}
